// MultiheadAttentionRoPE_1451698946549
// MI455X (gfx1250) — matrix operand order repaired and checked by reading, not compiled, not run on hardware
//
#include <hip/hip_runtime.h>
#include <hip/hip_bf16.h>

#ifndef __has_builtin
#define __has_builtin(x) 0
#endif

// Problem constants (from reference)
static constexpr int N_B  = 2;
static constexpr int N_L  = 8192;
static constexpr int N_D  = 1024;
static constexpr int N_H  = 16;
static constexpr int N_HD = 64;

typedef __attribute__((ext_vector_type(16))) __bf16   bf16x16;
typedef __attribute__((ext_vector_type(8)))  float    f32x8;
typedef __attribute__((ext_vector_type(4)))  unsigned u32x4;
typedef __attribute__((ext_vector_type(8)))  int      i32x8;
typedef __attribute__((ext_vector_type(4)))  int      i32x4;

// ---------------------------------------------------------------------------
// WMMA helpers (CDNA5 16x16x32 bf16, f32 accumulate)
// ---------------------------------------------------------------------------
static __device__ __forceinline__ f32x8 wmma_bf16(bf16x16 a, bf16x16 b, f32x8 c) {
  return __builtin_amdgcn_wmma_f32_16x16x32_bf16(false, a, false, b, (short)0, c,
                                                 false, false);
}

// A fragment: 16x32 (MxK) bf16, row-major source, stride lda elems.
// lane<16 -> row=lane, K = half*8+[0..7] and 16+half*8+[0..7]
static __device__ __forceinline__ bf16x16 load_a_frag(const __bf16* base, int lda) {
  int lane = threadIdx.x & 31;
  int r = lane & 15, half = lane >> 4;
  const __bf16* p0 = base + (size_t)r * lda + half * 8;
  bf16x16 a;
  ((uint4*)&a)[0] = *(const uint4*)p0;
  ((uint4*)&a)[1] = *(const uint4*)(p0 + 16);
  return a;
}

// B fragment: 32x16 (KxN) bf16 read from a transposed [N][K] source, stride ldt.
// lane -> N = lane&15 ; K = half*8+[0..7] and 16+half*8+[0..7]
static __device__ __forceinline__ bf16x16 load_b_frag_T(const __bf16* base, int ldt) {
  int lane = threadIdx.x & 31;
  int n = lane & 15, half = lane >> 4;
  const __bf16* p = base + (size_t)n * ldt + half * 8;
  bf16x16 b;
  ((uint4*)&b)[0] = *(const uint4*)p;
  ((uint4*)&b)[1] = *(const uint4*)(p + 16);
  return b;
}

// ---------------------------------------------------------------------------
// Tensor Data Mover: 2-D bf16 tile -> LDS (D# per ISA cdna5 §8.3/8.4)
// ---------------------------------------------------------------------------
#if __has_builtin(__builtin_amdgcn_tensor_load_to_lds)
#define HAVE_TDM 1
static __device__ __forceinline__ void tdm_load_2d_bf16(
    unsigned lds_off, const void* gptr, unsigned tile_x, unsigned tile_y,
    unsigned row_stride, unsigned tensor_x, unsigned tensor_y,
    unsigned pad_en, unsigned pad_interval_code, unsigned pad_amount_code) {
  unsigned long long ga = (unsigned long long)(uintptr_t)gptr;
  u32x4 g0 = {0u, 0u, 0u, 0u};
  g0[0] = 1u;                                            // count=1, user mode
  g0[1] = lds_off;                                       // lds_addr (bytes)
  g0[2] = (unsigned)ga;                                  // global_addr[31:0]
  g0[3] = ((unsigned)(ga >> 32) & 0x01ffffffu) | 0x80000000u; // [56:32] | type=2
  unsigned dw0 = (1u << 16)                              // data_size = 2 bytes
               | (pad_en << 20)
               | (pad_interval_code << 22)
               | (pad_amount_code << 25);
  unsigned dw1 = (tensor_x & 0xffffu) << 16;             // tensor_dim0[15:0]
  unsigned dw2 = (tensor_x >> 16) | ((tensor_y & 0xffffu) << 16);
  unsigned dw3 = (tensor_y >> 16) | ((tile_x & 0xffffu) << 16); // tile_dim0
  unsigned dw4 = tile_y & 0xffffu;                       // tile_dim1 (tile_dim2=0)
  unsigned dw5 = row_stride;                             // tensor_dim0_stride[31:0]
  i32x8 g1;
  g1[0]=(int)dw0; g1[1]=(int)dw1; g1[2]=(int)dw2; g1[3]=(int)dw3;
  g1[4]=(int)dw4; g1[5]=(int)dw5; g1[6]=0; g1[7]=0;
  i32x4 z4 = {0, 0, 0, 0};
#if __clang_major__ >= 23
  i32x8 z8 = {0, 0, 0, 0, 0, 0, 0, 0};
  __builtin_amdgcn_tensor_load_to_lds(g0, g1, z4, z4, z8, 0);
#else
  __builtin_amdgcn_tensor_load_to_lds(g0, g1, z4, z4, 0);
#endif
}
#else
#define HAVE_TDM 0
#endif

// ---------------------------------------------------------------------------
// Elementwise kernels
// ---------------------------------------------------------------------------
__global__ void cast_f32_bf16(const float* __restrict__ s, __bf16* __restrict__ d,
                              size_t n) {
  size_t i = (size_t)blockIdx.x * blockDim.x + threadIdx.x;
  size_t stride = (size_t)gridDim.x * blockDim.x;
  for (; i < n; i += stride) d[i] = (__bf16)s[i];
}

// W: [K][N] f32 (row-major)  ->  Wt: [N][K] bf16 (transposed)
__global__ void cast_transpose_bf16(const float* __restrict__ W,
                                    __bf16* __restrict__ Wt, int N, int K) {
  int i = blockIdx.x * blockDim.x + threadIdx.x;
  if (i >= N * K) return;
  int k = i % K, n = i / K;  // adjacent threads write adjacent k: coalesced out
  Wt[(size_t)n * K + k] = (__bf16)W[(size_t)k * N + n];
}

__global__ void rope_tables(float* __restrict__ sint, float* __restrict__ cost) {
  int i = blockIdx.x * blockDim.x + threadIdx.x;
  if (i >= N_L * N_HD) return;
  int l = i >> 6, dd = i & 63;
  float ex = (float)(dd & ~1) / (float)N_HD;
  float freq = __powf(10000.0f, -ex);
  float ang = (float)l * freq;
  sint[i] = __sinf(ang);
  cost[i] = __cosf(ang);
}

// In-place RoPE: each thread owns one head's (d, d+32) pair -> race-free.
__global__ void rope_apply(__bf16* __restrict__ qk, const float* __restrict__ sint,
                           const float* __restrict__ cost) {
  int i = blockIdx.x * blockDim.x + threadIdx.x;  // over N_B*N_L*N_H*32
  if (i >= N_B * N_L * N_H * 32) return;
  int p = i & 31;
  int h = (i >> 5) & (N_H - 1);
  int l = (i >> 9) & (N_L - 1);
  int b = i >> 22;
  size_t base = ((size_t)(b * N_L + l)) * N_D + h * N_HD;
  float v1 = (float)qk[base + p];
  float v2 = (float)qk[base + p + 32];
  float s1 = sint[l * N_HD + p],      c1 = cost[l * N_HD + p];
  float s2 = sint[l * N_HD + p + 32], c2 = cost[l * N_HD + p + 32];
  qk[base + p]      = (__bf16)(v1 * c1 - v2 * s1);
  qk[base + p + 32] = (__bf16)(v2 * c2 + v1 * s2);
}

// ---------------------------------------------------------------------------
// WMMA GEMM v2: out[M x 1024] = A[M x 1024](bf16) @ W + bias,
// W given pre-transposed as Wt[N][K] bf16.  block = 256 (8 waves), 1-D grid
// of M/16 blocks.  Wave w owns a 16x128 strip (8 accumulators); the block's
// 16x1024 A-slab is staged in LDS once (TDM if available), then the k-loop
// runs with ZERO barriers: per k-step = 1 A-frag + 8 B-frags + 8 WMMAs.
// mode 0: bf16 out ; mode 1: fp32 out.
// ---------------------------------------------------------------------------
__global__ __launch_bounds__(256) void gemm_bf16_v2(
    const __bf16* __restrict__ A, const __bf16* __restrict__ Wt,
    const float* __restrict__ bias, __bf16* __restrict__ outb,
    float* __restrict__ outf, int M, int N, int K, int mode) {
  __shared__ __align__(16) __bf16 ldsA[16][N_D];  // 32 KB

  int wave = threadIdx.x >> 5;
  int lane = threadIdx.x & 31;
  int half = lane >> 4;
  int m0 = blockIdx.x * 16;
  int n0 = wave * 128;

  // ---- stage A slab (16 rows x K) into LDS ----
#if HAVE_TDM
  if (wave == 0) {
    tdm_load_2d_bf16((unsigned)(uintptr_t)&ldsA[0][0],
                     A + (size_t)m0 * K, /*tile_x=*/K, /*tile_y=*/16,
                     /*row_stride=*/K, /*tensor_x=*/K, /*tensor_y=*/M - m0,
                     /*pad_en=*/0, 0, 0);
    __builtin_amdgcn_s_wait_tensorcnt(0);
  }
#else
  {
    int row = threadIdx.x >> 4;          // 0..15
    int c0  = (threadIdx.x & 15) * 64;   // 64 elems per thread
    const __bf16* ap = A + (size_t)(m0 + row) * K + c0;
#pragma unroll
    for (int j = 0; j < 8; ++j)
      *(uint4*)&ldsA[row][c0 + j * 8] = *(const uint4*)(ap + j * 8);
  }
#endif
  __syncthreads();

  f32x8 acc[8];
#pragma unroll
  for (int t = 0; t < 8; ++t) acc[t] = (f32x8){};

  for (int k0 = 0; k0 < K; k0 += 32) {
    bf16x16 a = load_a_frag(&ldsA[0][k0], N_D);
#pragma unroll
    for (int t = 0; t < 8; ++t) {
      bf16x16 b = load_b_frag_T(Wt + (size_t)(n0 + t * 16) * K + k0, K);
      acc[t] = wmma_bf16(a, b, acc[t]);
    }
  }

  // ---- epilogue ----
  if (mode == 0) {
#pragma unroll
    for (int t = 0; t < 8; ++t) {
      int n = n0 + t * 16 + (lane & 15);
      float bv = bias[n];
#pragma unroll
      for (int v = 0; v < 8; ++v)
        outb[(size_t)(m0 + v + half * 8) * N + n] = (__bf16)(acc[t][v] + bv);
    }
  } else {
#pragma unroll
    for (int t = 0; t < 8; ++t) {
      int n = n0 + t * 16 + (lane & 15);
      float bv = bias[n];
#pragma unroll
      for (int v = 0; v < 8; ++v)
        outf[(size_t)(m0 + v + half * 8) * N + n] = acc[t][v] + bv;
    }
  }
}

// ---------------------------------------------------------------------------
// Flash attention (non-causal).  grid = (B*H, L/128), block = 256 (8 waves).
// Wave w owns query rows [blockIdx.y*128 + w*16, +16).  Per 64-key tile:
// K tile staged by TDM (with HW padding to [64][72]), V tile transposed with
// key-pair-packed b32 LDS stores, S = Q@K^T (8 WMMAs), online softmax,
// P relayout through wave-private LDS (s_wait_dscnt), O += P@V (8 WMMAs).
// ---------------------------------------------------------------------------
__global__ __launch_bounds__(256) void attn_fwd(
    const __bf16* __restrict__ Q, const __bf16* __restrict__ Kc,
    const __bf16* __restrict__ V, __bf16* __restrict__ O) {
  __shared__ __align__(16) __bf16 ldsK[64][72];     // [key][hd]
  __shared__ __align__(16) __bf16 ldsVt[64][72];    // [hd][key]
  __shared__ __align__(16) __bf16 ldsP[8][16][72];  // per-wave P [row][key]

  int wave = threadIdx.x >> 5;
  int lane = threadIdx.x & 31;
  int half = lane >> 4;
  int b = blockIdx.x / N_H;
  int h = blockIdx.x % N_H;
  int m0 = blockIdx.y * 128 + wave * 16;

  const float sm_scale = 0.125f;  // 1/sqrt(64)

  const __bf16* qbase = Q + ((size_t)(b * N_L + m0)) * N_D + h * N_HD;
  bf16x16 aq0 = load_a_frag(qbase + 0, N_D);
  bf16x16 aq1 = load_a_frag(qbase + 32, N_D);

  f32x8 oacc[4];
#pragma unroll
  for (int n = 0; n < 4; ++n) oacc[n] = (f32x8){};
  float mrow[8], lrow[8];
#pragma unroll
  for (int v = 0; v < 8; ++v) { mrow[v] = -3.0e38f; lrow[v] = 0.0f; }

  for (int kt = 0; kt < N_L / 64; ++kt) {
    // ---- V tile transposed into ldsVt, two keys packed per b32 store ----
    {
      int kp  = threadIdx.x >> 3;        // 0..31 -> keys 2kp, 2kp+1
      int hd0 = (threadIdx.x & 7) * 8;   // 0,8,...,56
      const __bf16* va =
          V + ((size_t)(b * N_L + kt * 64 + 2 * kp)) * N_D + h * N_HD + hd0;
      uint4 ra = *(const uint4*)va;
      uint4 rb = *(const uint4*)(va + N_D);
      unsigned aw[4] = {ra.x, ra.y, ra.z, ra.w};
      unsigned bw[4] = {rb.x, rb.y, rb.z, rb.w};
#pragma unroll
      for (int i = 0; i < 8; ++i) {
        unsigned lo = (i & 1) ? (aw[i >> 1] >> 16) : (aw[i >> 1] & 0xffffu);
        unsigned hi = (i & 1) ? (bw[i >> 1] & 0xffff0000u) : (bw[i >> 1] << 16);
        *(unsigned*)&ldsVt[hd0 + i][2 * kp] = lo | hi;
      }
    }
    // ---- K tile: TDM with hardware padding (row stride 72) ----
#if HAVE_TDM
    if (wave == 0) {
      tdm_load_2d_bf16((unsigned)(uintptr_t)&ldsK[0][0],
                       Kc + ((size_t)(b * N_L + kt * 64)) * N_D + h * N_HD,
                       /*tile_x=*/64, /*tile_y=*/64, /*row_stride=*/N_D,
                       /*tensor_x=*/N_D, /*tensor_y=*/64,
                       /*pad_en=*/1, /*interval: 32 dwords=*/4, /*amount: 4 dwords=*/3);
      __builtin_amdgcn_s_wait_tensorcnt(0);
    }
#else
    {
      int krow = threadIdx.x >> 2;
      int kcol = (threadIdx.x & 3) * 16;
      const __bf16* kp =
          Kc + ((size_t)(b * N_L + kt * 64 + krow)) * N_D + h * N_HD + kcol;
      *(uint4*)&ldsK[krow][kcol]     = *(const uint4*)kp;
      *(uint4*)&ldsK[krow][kcol + 8] = *(const uint4*)(kp + 8);
    }
#endif
    __syncthreads();

    // ---- S = Q @ K^T : 4 N-tiles x 2 k-steps ----
    f32x8 s[4];
#pragma unroll
    for (int n = 0; n < 4; ++n) {
      f32x8 c = {};
      c = wmma_bf16(aq0, load_b_frag_T(&ldsK[n * 16][0], 72), c);
      c = wmma_bf16(aq1, load_b_frag_T(&ldsK[n * 16][32], 72), c);
      s[n] = c;
    }

    // ---- online softmax ----
    float mnew[8];
#pragma unroll
    for (int v = 0; v < 8; ++v) {
      float mx = fmaxf(fmaxf(s[0][v], s[1][v]), fmaxf(s[2][v], s[3][v])) * sm_scale;
      for (int off = 1; off < 16; off <<= 1)
        mx = fmaxf(mx, __shfl_xor(mx, off, 32));
      mnew[v] = fmaxf(mrow[v], mx);
    }
#pragma unroll
    for (int v = 0; v < 8; ++v) {
      float corr = __expf(mrow[v] - mnew[v]);
      mrow[v] = mnew[v];
      lrow[v] *= corr;
      float psum = 0.0f;
#pragma unroll
      for (int n = 0; n < 4; ++n) oacc[n][v] *= corr;
#pragma unroll
      for (int n = 0; n < 4; ++n) {
        float p = __expf(s[n][v] * sm_scale - mnew[v]);
        s[n][v] = p;
        psum += p;
      }
      for (int off = 1; off < 16; off <<= 1)
        psum += __shfl_xor(psum, off, 32);
      lrow[v] += psum;
    }

    // ---- P : C-layout -> A-layout via wave-private LDS ----
#pragma unroll
    for (int n = 0; n < 4; ++n)
#pragma unroll
      for (int v = 0; v < 8; ++v)
        ldsP[wave][v + half * 8][n * 16 + (lane & 15)] = (__bf16)s[n][v];
    asm volatile("s_wait_dscnt 0" ::: "memory");

    bf16x16 ap0 = load_a_frag(&ldsP[wave][0][0], 72);
    bf16x16 ap1 = load_a_frag(&ldsP[wave][0][32], 72);

    // ---- O += P @ V : 4 hd-tiles x 2 k-steps ----
#pragma unroll
    for (int n = 0; n < 4; ++n) {
      oacc[n] = wmma_bf16(ap0, load_b_frag_T(&ldsVt[n * 16][0], 72), oacc[n]);
      oacc[n] = wmma_bf16(ap1, load_b_frag_T(&ldsVt[n * 16][32], 72), oacc[n]);
    }
    __syncthreads();  // protect ldsK/ldsVt for next tile
  }

  // ---- epilogue ----
#pragma unroll
  for (int n = 0; n < 4; ++n) {
#pragma unroll
    for (int v = 0; v < 8; ++v) {
      int m = m0 + v + half * 8;
      int c = h * N_HD + n * 16 + (lane & 15);
      O[((size_t)(b * N_L + m)) * N_D + c] = (__bf16)(oacc[n][v] / lrow[v]);
    }
  }
}

// ---------------------------------------------------------------------------
// Host launcher
// ---------------------------------------------------------------------------
extern "C" void kernel_launch(void* const* d_in, const int* in_sizes, int n_in,
                              void* d_out, int out_size, void* d_ws, size_t ws_size,
                              hipStream_t stream) {
  (void)in_sizes; (void)n_in; (void)out_size; (void)ws_size;
  const float* x  = (const float*)d_in[0];
  const float* Wq = (const float*)d_in[1];
  const float* bq = (const float*)d_in[2];
  const float* Wk = (const float*)d_in[3];
  const float* bk = (const float*)d_in[4];
  const float* Wv = (const float*)d_in[5];
  const float* bv = (const float*)d_in[6];
  const float* Wo = (const float*)d_in[7];
  const float* bo = (const float*)d_in[8];

  char* w = (char*)d_ws;
  auto alloc = [&](size_t bytes) -> char* {
    char* p = w;
    w += (bytes + 255) & ~(size_t)255;
    return p;
  };
  const size_t NX = (size_t)N_B * N_L * N_D;
  const size_t NW = (size_t)N_D * N_D;
  __bf16* xb   = (__bf16*)alloc(NX * 2);
  __bf16* WqT  = (__bf16*)alloc(NW * 2);
  __bf16* WkT  = (__bf16*)alloc(NW * 2);
  __bf16* WvT  = (__bf16*)alloc(NW * 2);
  __bf16* WoT  = (__bf16*)alloc(NW * 2);
  __bf16* Qb   = (__bf16*)alloc(NX * 2);
  __bf16* Kb   = (__bf16*)alloc(NX * 2);
  __bf16* Vb   = (__bf16*)alloc(NX * 2);
  __bf16* Ab   = (__bf16*)alloc(NX * 2);
  float*  sint = (float*)alloc((size_t)N_L * N_HD * 4);
  float*  cost = (float*)alloc((size_t)N_L * N_HD * 4);

  // 1) casts / transposes / tables
  cast_f32_bf16<<<2048, 256, 0, stream>>>(x, xb, NX);
  int gtw = (int)((NW + 255) / 256);
  cast_transpose_bf16<<<gtw, 256, 0, stream>>>(Wq, WqT, N_D, N_D);
  cast_transpose_bf16<<<gtw, 256, 0, stream>>>(Wk, WkT, N_D, N_D);
  cast_transpose_bf16<<<gtw, 256, 0, stream>>>(Wv, WvT, N_D, N_D);
  cast_transpose_bf16<<<gtw, 256, 0, stream>>>(Wo, WoT, N_D, N_D);
  rope_tables<<<(N_L * N_HD + 255) / 256, 256, 0, stream>>>(sint, cost);

  // 2) Q/K/V projections
  int M = N_B * N_L;
  gemm_bf16_v2<<<M / 16, 256, 0, stream>>>(xb, WqT, bq, Qb, nullptr, M, N_D, N_D, 0);
  gemm_bf16_v2<<<M / 16, 256, 0, stream>>>(xb, WkT, bk, Kb, nullptr, M, N_D, N_D, 0);
  gemm_bf16_v2<<<M / 16, 256, 0, stream>>>(xb, WvT, bv, Vb, nullptr, M, N_D, N_D, 0);

  // 3) RoPE on Q, K
  int nr = N_B * N_L * N_H * 32;
  rope_apply<<<(nr + 255) / 256, 256, 0, stream>>>(Qb, sint, cost);
  rope_apply<<<(nr + 255) / 256, 256, 0, stream>>>(Kb, sint, cost);

  // 4) Flash attention
  dim3 ga(N_B * N_H, N_L / 128);
  attn_fwd<<<ga, 256, 0, stream>>>(Qb, Kb, Vb, Ab);

  // 5) Output projection (fp32 + bias -> d_out)
  gemm_bf16_v2<<<M / 16, 256, 0, stream>>>(Ab, WoT, bo, nullptr, (float*)d_out,
                                           M, N_D, N_D, 1);
}
